// AssignAttention_52621939311054
// MI455X (gfx1250) — matrix operand order repaired and checked by reading, not compiled, not run on hardware
//
#include <hip/hip_runtime.h>
#include <stdint.h>

// Problem constants (from reference)
#define BATCH 2
#define SEQ   2048
#define CH    1024
#define HEADS 8
#define HDIM  128
#define BH    (BATCH*HEADS)     // 16
#define MROWS (BATCH*SEQ)       // 4096

typedef __attribute__((ext_vector_type(16))) __bf16 v16bf;
typedef __attribute__((ext_vector_type(8)))  float  v8f;

struct __attribute__((aligned(16))) U128 { unsigned int a, b, c, d; };
union FragU { U128 q[2]; v16bf v; };

#define WMMA_BF16(Cacc, Afrag, Bfrag) \
  __builtin_amdgcn_wmma_f32_16x16x32_bf16(false, (Afrag), false, (Bfrag), (short)0, (Cacc), false, false)

__device__ __forceinline__ unsigned short f2bf(float f) {
  unsigned int u = __float_as_uint(f);
  u += 0x7FFFu + ((u >> 16) & 1u);          // round to nearest even
  return (unsigned short)(u >> 16);
}
__device__ __forceinline__ float bf2f(unsigned short h) {
  return __uint_as_float(((unsigned int)h) << 16);
}

// A fragment (16x32 bf16, rows = M, row-major [M,K] source).
// ISA layout: lane L holds row m=L&15; halves i=0..7 -> k = kc + hi*8 + i,
//             halves i=8..15 -> k = kc + 16 + hi*8 + (i-8). Two b128 loads.
__device__ __forceinline__ v16bf load_frag_A(const unsigned short* base, int stride,
                                             int m0, int kc, int lane) {
  int m = m0 + (lane & 15);
  int hi = lane >> 4;
  const unsigned short* p = base + (size_t)m * stride + kc + hi * 8;
  FragU f;
  f.q[0] = *(const U128*)(p);
  f.q[1] = *(const U128*)(p + 16);
  return f.v;
}

// B fragment (32x16 bf16, cols = N) loaded from a TRANSPOSED source [N,K]:
// lane L holds col n=L&15; halves i=0..7 -> k = kc + hi*8 + i, i=8..15 -> k = kc + 16 + hi*8 + (i-8).
__device__ __forceinline__ v16bf load_frag_B(const unsigned short* baseT, int stride,
                                             int n0, int kc, int lane) {
  int n = n0 + (lane & 15);
  int hi = lane >> 4;
  const unsigned short* p = baseT + (size_t)n * stride + kc + hi * 8;
  FragU f;
  f.q[0] = *(const U128*)(p);
  f.q[1] = *(const U128*)(p + 16);
  return f.v;
}

// ---------- conversion kernels ----------
__global__ void __launch_bounds__(256) cvt_kernel(const float* in, unsigned short* out, int n) {
  int i = blockIdx.x * blockDim.x + threadIdx.x;
  if (i < n) out[i] = f2bf(in[i]);
}

// Wt[n*CH + k] = W[k*CH + n]  (transpose + bf16)
__global__ void __launch_bounds__(256) tcvt_kernel(const float* W, unsigned short* Wt) {
  int i = blockIdx.x * blockDim.x + threadIdx.x;
  int n = i >> 10, k = i & (CH - 1);
  Wt[i] = f2bf(W[(size_t)k * CH + n]);
}

// ---------- bf16 WMMA GEMM: Out[M,N] = A[M,K] * Bt[N,K]^T + bias ----------
// Wave tile 32x64 (8 f32 accumulators); block = 4 waves -> 64x128 tile.
// mode 0: store bf16 to [B,H,Nseq,HD] head layout (for q/k/v)
// mode 1: store f32 row-major (final projection into d_out)
__device__ __forceinline__ void store_tile(const v8f& c, int tm0, int tn0, float bv,
                                           unsigned short* dst_bf, float* dst_f,
                                           int Nd, int mode, int lane) {
  int hi = lane >> 4, ln = lane & 15;
  int n = tn0 + ln;
#pragma unroll
  for (int r = 0; r < 8; ++r) {
    int m = tm0 + r + hi * 8;
    float val = c[r] + bv;
    if (mode == 0) {
      int b = m >> 11, ns = m & (SEQ - 1);
      int h = n >> 7, hd = n & (HDIM - 1);
      dst_bf[(((size_t)b * HEADS + h) * SEQ + ns) * HDIM + hd] = f2bf(val);
    } else {
      dst_f[(size_t)m * Nd + n] = val;
    }
  }
}

__global__ void __launch_bounds__(128) gemm_bf16(const unsigned short* A,
                                                 const unsigned short* Bt,
                                                 const float* bias,
                                                 unsigned short* dst_bf, float* dst_f,
                                                 int Nd, int Kd, int mode) {
  int lane = threadIdx.x & 31;
  int wave = threadIdx.x >> 5;
  int wm = wave >> 1, wn = wave & 1;
  int mBase = blockIdx.y * 64 + wm * 32;
  int nBase = blockIdx.x * 128 + wn * 64;

  v8f cA0 = {}, cA1 = {}, cA2 = {}, cA3 = {};
  v8f cB0 = {}, cB1 = {}, cB2 = {}, cB3 = {};
  for (int kc = 0; kc < Kd; kc += 32) {
    // unconditional speculative prefetch of the next A chunk
    __builtin_prefetch(A + (size_t)(mBase + (lane & 15)) * Kd + kc + 32, 0, 1);
    v16bf a0 = load_frag_A(A, Kd, mBase,      kc, lane);
    v16bf a1 = load_frag_A(A, Kd, mBase + 16, kc, lane);
    v16bf b0 = load_frag_B(Bt, Kd, nBase,      kc, lane);
    v16bf b1 = load_frag_B(Bt, Kd, nBase + 16, kc, lane);
    v16bf b2 = load_frag_B(Bt, Kd, nBase + 32, kc, lane);
    v16bf b3 = load_frag_B(Bt, Kd, nBase + 48, kc, lane);
    cA0 = WMMA_BF16(cA0, a0, b0);
    cA1 = WMMA_BF16(cA1, a0, b1);
    cA2 = WMMA_BF16(cA2, a0, b2);
    cA3 = WMMA_BF16(cA3, a0, b3);
    cB0 = WMMA_BF16(cB0, a1, b0);
    cB1 = WMMA_BF16(cB1, a1, b1);
    cB2 = WMMA_BF16(cB2, a1, b2);
    cB3 = WMMA_BF16(cB3, a1, b3);
  }
  int ln = lane & 15;
  float bv0 = bias[nBase + ln];
  float bv1 = bias[nBase + 16 + ln];
  float bv2 = bias[nBase + 32 + ln];
  float bv3 = bias[nBase + 48 + ln];
  store_tile(cA0, mBase,      nBase,      bv0, dst_bf, dst_f, Nd, mode, lane);
  store_tile(cA1, mBase,      nBase + 16, bv1, dst_bf, dst_f, Nd, mode, lane);
  store_tile(cA2, mBase,      nBase + 32, bv2, dst_bf, dst_f, Nd, mode, lane);
  store_tile(cA3, mBase,      nBase + 48, bv3, dst_bf, dst_f, Nd, mode, lane);
  store_tile(cB0, mBase + 16, nBase,      bv0, dst_bf, dst_f, Nd, mode, lane);
  store_tile(cB1, mBase + 16, nBase + 16, bv1, dst_bf, dst_f, Nd, mode, lane);
  store_tile(cB2, mBase + 16, nBase + 32, bv2, dst_bf, dst_f, Nd, mode, lane);
  store_tile(cB3, mBase + 16, nBase + 48, bv3, dst_bf, dst_f, Nd, mode, lane);
}

// ---------- top-k pass: per (b,h,s), top-4 over n of rawT[s,n] = k[s]·q[n] ----------
// 4 waves/block; each wave owns a 16-row s-strip, streams all n tiles via WMMA.
__global__ void __launch_bounds__(128) topk_kernel(const unsigned short* kh,
                                                   const unsigned short* qh,
                                                   int* topk) {
  __shared__ float sm[4][16][16];
  int lane = threadIdx.x & 31;
  int wave = threadIdx.x >> 5;
  int gid = blockIdx.x;              // BH * (SEQ/64) = 512 blocks
  int bh = gid >> 5;                 // 32 blocks per (b,h)
  int s0 = ((gid & 31) << 6) + wave * 16;
  const unsigned short* kb = kh + (size_t)bh * SEQ * HDIM;
  const unsigned short* qb = qh + (size_t)bh * SEQ * HDIM;

  // A = k rows [s0..s0+15] x HD=128 (resident: 4 K-chunks)
  v16bf a0 = load_frag_A(kb, HDIM, s0,  0, lane);
  v16bf a1 = load_frag_A(kb, HDIM, s0, 32, lane);
  v16bf a2 = load_frag_A(kb, HDIM, s0, 64, lane);
  v16bf a3 = load_frag_A(kb, HDIM, s0, 96, lane);

  float t0 = -3.0e38f, t1 = -3.0e38f, t2 = -3.0e38f, t3 = -3.0e38f;
  int   i0 = 0, i1 = 0, i2 = 0, i3 = 0;
  int hi = lane >> 4, ln = lane & 15;

  for (int nt = 0; nt < SEQ / 16; ++nt) {
    int n0 = nt * 16;
    v16bf b0 = load_frag_B(qb, HDIM, n0,  0, lane);
    v16bf b1 = load_frag_B(qb, HDIM, n0, 32, lane);
    v16bf b2 = load_frag_B(qb, HDIM, n0, 64, lane);
    v16bf b3 = load_frag_B(qb, HDIM, n0, 96, lane);
    v8f acc = {};
    acc = WMMA_BF16(acc, a0, b0);
    acc = WMMA_BF16(acc, a1, b1);
    acc = WMMA_BF16(acc, a2, b2);
    acc = WMMA_BF16(acc, a3, b3);
#pragma unroll
    for (int r = 0; r < 8; ++r) sm[wave][r + hi * 8][ln] = acc[r];
    __syncthreads();
    if (lane < 16) {
#pragma unroll 4
      for (int j = 0; j < 16; ++j) {
        float v = sm[wave][lane][j];
        if (v > t3) {                       // strict '>' => lower index wins ties
          int n = n0 + j;
          if (v > t0)      { t3=t2; i3=i2; t2=t1; i2=i1; t1=t0; i1=i0; t0=v; i0=n; }
          else if (v > t1) { t3=t2; i3=i2; t2=t1; i2=i1; t1=v; i1=n; }
          else if (v > t2) { t3=t2; i3=i2; t2=v; i2=n; }
          else             { t3=v; i3=n; }
        }
      }
    }
    __syncthreads();
  }
  if (lane < 16) {
    int s = s0 + lane;
    int* o = topk + ((size_t)bh * SEQ + s) * 4;
    o[0] = i0; o[1] = i1; o[2] = i2; o[3] = i3;
  }
}

// ---------- scatter: counts + S1v[n] = sum over unmasked s selecting n of v[s] ----------
__global__ void __launch_bounds__(256) scatter_kernel(const int* topk, const int* mask,
                                                      const unsigned short* vh,
                                                      unsigned int* cntAll,
                                                      unsigned int* cntUm,
                                                      float* S1v) {
  int lane = threadIdx.x & 31;
  int item = blockIdx.x * 8 + (threadIdx.x >> 5);   // item = bh*SEQ + s
  int bh = item >> 11;
  int s  = item & (SEQ - 1);
  int b  = bh >> 3;
  int idx[4];
#pragma unroll
  for (int j = 0; j < 4; ++j) idx[j] = topk[(size_t)item * 4 + j];
  int msk = mask[b * SEQ + s];
  if (lane == 0) {
#pragma unroll
    for (int j = 0; j < 4; ++j) {
      atomicAdd(&cntAll[bh * SEQ + idx[j]], 1u);
      if (msk) atomicAdd(&cntUm[bh * SEQ + idx[j]], 1u);
    }
  }
  if (msk) {
    const unsigned short* vr = vh + ((size_t)bh * SEQ + s) * HDIM;
    float vv[4];
#pragma unroll
    for (int t = 0; t < 4; ++t) vv[t] = bf2f(vr[lane + 32 * t]);
#pragma unroll
    for (int j = 0; j < 4; ++j) {
      float* d = S1v + ((size_t)bh * SEQ + idx[j]) * HDIM;
#pragma unroll
      for (int t = 0; t < 4; ++t) atomicAdd(&d[lane + 32 * t], vv[t]);
    }
  }
}

// ---------- per-(b,h) unmasked V sum + unmasked column count ----------
__global__ void __launch_bounds__(128) vsum_kernel(const unsigned short* vh, const int* mask,
                                                   float* Vsum, unsigned int* Ucnt) {
  int bh = blockIdx.x;
  int hd = threadIdx.x;
  int b = bh >> 3;
  const unsigned short* vb = vh + (size_t)bh * SEQ * HDIM;
  const int* mb = mask + b * SEQ;
  float acc = 0.f; unsigned int cnt = 0;
  for (int s = 0; s < SEQ; ++s) {
    if (mb[s]) { acc += bf2f(vb[(size_t)s * HDIM + hd]); cnt++; }
  }
  Vsum[bh * HDIM + hd] = acc;
  if (hd == 0) Ucnt[bh] = cnt;
}

// ---------- closed-form two-value softmax + attn@v assembly ----------
__global__ void __launch_bounds__(256) attnout_kernel(const unsigned int* cntAll,
                                                      const unsigned int* cntUm,
                                                      const unsigned int* Ucnt,
                                                      const float* S1v, const float* Vsum,
                                                      unsigned short* aout) {
  int lane = threadIdx.x & 31;
  int item = blockIdx.x * 8 + (threadIdx.x >> 5);   // item = bh*SEQ + n
  int bh = item >> 11;
  int n  = item & (SEQ - 1);
  int b = bh >> 3, h = bh & 7;
  float w  = 1.0f / ((float)cntAll[bh * SEQ + n] + 1.0f);   // EPS = 1.0
  float c1 = (float)cntUm[bh * SEQ + n];
  float c0 = (float)Ucnt[bh] - c1;
  float m  = (c1 > 0.f) ? w : 0.f;                          // masked cols -> exp==0
  float e1 = __expf(w - m);
  float e0 = __expf(-m);
  float Z  = c1 * e1 + c0 * e0;
  float p1 = (Z > 0.f) ? e1 / Z : 0.f;
  float p0 = (Z > 0.f) ? e0 / Z : 0.f;
  const float* s1 = S1v + ((size_t)bh * SEQ + n) * HDIM;
  const float* vs = Vsum + bh * HDIM;
  unsigned short* dst = aout + ((size_t)b * SEQ + n) * CH + h * HDIM;
#pragma unroll
  for (int t = 0; t < 4; ++t) {
    int hd = lane + 32 * t;
    float sv = s1[hd];
    dst[hd] = f2bf(p1 * sv + p0 * (vs[hd] - sv));
  }
}

extern "C" void kernel_launch(void* const* d_in, const int* in_sizes, int n_in,
                              void* d_out, int out_size, void* d_ws, size_t ws_size,
                              hipStream_t stream) {
  const float* query = (const float*)d_in[0];
  const int*   mask  = (const int*)  d_in[1];
  const float* Wq = (const float*)d_in[2]; const float* bq = (const float*)d_in[3];
  const float* Wk = (const float*)d_in[4]; const float* bk = (const float*)d_in[5];
  const float* Wv = (const float*)d_in[6]; const float* bv = (const float*)d_in[7];
  const float* Wp = (const float*)d_in[8]; const float* bp = (const float*)d_in[9];
  float* out = (float*)d_out;

  char* ws = (char*)d_ws;
  size_t off = 0;
  auto alloc = [&](size_t bytes) -> char* {
    char* p = ws + off; off += (bytes + 255) & ~(size_t)255; return p;
  };
  unsigned short* qryb = (unsigned short*)alloc((size_t)MROWS * CH * 2);   // also reused as aout
  unsigned short* Wqt  = (unsigned short*)alloc((size_t)CH * CH * 2);
  unsigned short* Wkt  = (unsigned short*)alloc((size_t)CH * CH * 2);
  unsigned short* Wvt  = (unsigned short*)alloc((size_t)CH * CH * 2);
  unsigned short* Wpt  = (unsigned short*)alloc((size_t)CH * CH * 2);
  unsigned short* qh   = (unsigned short*)alloc((size_t)BH * SEQ * HDIM * 2);
  unsigned short* kh   = (unsigned short*)alloc((size_t)BH * SEQ * HDIM * 2);
  unsigned short* vh   = (unsigned short*)alloc((size_t)BH * SEQ * HDIM * 2);
  int*            tpk  = (int*)          alloc((size_t)BH * SEQ * 4 * 4);
  unsigned int*   cntAll = (unsigned int*)alloc((size_t)BH * SEQ * 4);
  unsigned int*   cntUm  = (unsigned int*)alloc((size_t)BH * SEQ * 4);
  unsigned int*   Ucnt   = (unsigned int*)alloc(BH * 4);
  float*          S1v    = (float*)alloc((size_t)BH * SEQ * HDIM * 4);
  float*          Vsum   = (float*)alloc((size_t)BH * HDIM * 4);

  // 1) bf16 conversions (+ weight transposes so B-operands load contiguously)
  cvt_kernel<<<(MROWS * CH) / 256, 256, 0, stream>>>(query, qryb, MROWS * CH);
  tcvt_kernel<<<(CH * CH) / 256, 256, 0, stream>>>(Wq, Wqt);
  tcvt_kernel<<<(CH * CH) / 256, 256, 0, stream>>>(Wk, Wkt);
  tcvt_kernel<<<(CH * CH) / 256, 256, 0, stream>>>(Wv, Wvt);
  tcvt_kernel<<<(CH * CH) / 256, 256, 0, stream>>>(Wp, Wpt);

  // 2) q/k/v projections (WMMA), stored in head layout
  dim3 gg(CH / 128, MROWS / 64);
  gemm_bf16<<<gg, 128, 0, stream>>>(qryb, Wqt, bq, qh, nullptr, CH, CH, 0);
  gemm_bf16<<<gg, 128, 0, stream>>>(qryb, Wkt, bk, kh, nullptr, CH, CH, 0);
  gemm_bf16<<<gg, 128, 0, stream>>>(qryb, Wvt, bv, vh, nullptr, CH, CH, 0);

  // 3) per-column top-4 of q·k^T (WMMA score tiles, streaming top-k)
  topk_kernel<<<BH * (SEQ / 64), 128, 0, stream>>>(kh, qh, tpk);

  // 4) structured softmax + attn@v (scatter form)
  hipMemsetAsync(cntAll, 0, (size_t)BH * SEQ * 4, stream);
  hipMemsetAsync(cntUm,  0, (size_t)BH * SEQ * 4, stream);
  hipMemsetAsync(S1v,    0, (size_t)BH * SEQ * HDIM * 4, stream);
  vsum_kernel<<<BH, 128, 0, stream>>>(vh, mask, Vsum, Ucnt);
  scatter_kernel<<<(BH * SEQ) / 8, 256, 0, stream>>>(tpk, mask, vh, cntAll, cntUm, S1v);

  unsigned short* aout = qryb;  // reuse: qryb no longer needed after the 3 projections
  attnout_kernel<<<(BH * SEQ) / 8, 256, 0, stream>>>(cntAll, cntUm, Ucnt, S1v, Vsum, aout);

  // 5) final projection -> f32 d_out
  gemm_bf16<<<gg, 128, 0, stream>>>(aout, Wpt, bp, nullptr, out, CH, CH, 1);
}
